// GeneratorBlock_19292993094050
// MI455X (gfx1250) — compile-verified, matrix operand order checked by reading, not run on hardware
//
#include <hip/hip_runtime.h>
#include <stdint.h>

typedef __attribute__((ext_vector_type(16))) __bf16 v16bf;
typedef __attribute__((ext_vector_type(8)))  __bf16 v8bf;
typedef __attribute__((ext_vector_type(8)))  float  v8f;
typedef __attribute__((ext_vector_type(4)))  unsigned int v4u;
typedef __attribute__((ext_vector_type(8)))  unsigned int v8u;

#define B_  16
#define C_  512
#define H_  32
#define W_  32
#define HW_ 1024
#define O_  512
#define KK_ 4608      // 9*512 ; k-index = j*512 + c (tap-major)
#define NKC 144       // 4608 / 32
#define LDSTRIDE 40   // LDS row stride in bf16 elements (64B payload + 16B pad)

struct Halves { v8bf lo; v8bf hi; };
union Frag { v16bf v; Halves h; };

// ---------------- small pre-pass kernels ----------------

__global__ __launch_bounds__(256) void k_style(const float* __restrict__ w_embs,
                                               const float* __restrict__ style_w,
                                               const float* __restrict__ style_b,
                                               float* __restrict__ mod) {
  int t = blockIdx.x * blockDim.x + threadIdx.x;
  if (t >= B_ * C_) return;
  int b = t >> 9, c = t & (C_ - 1);
  const float4* we = (const float4*)(w_embs + (size_t)b * C_);
  const float4* sw = (const float4*)(style_w + (size_t)c * C_);
  float s = 0.f;
  for (int i = 0; i < C_ / 4; ++i) {
    float4 u = we[i], v = sw[i];
    s = fmaf(u.x, v.x, s); s = fmaf(u.y, v.y, s);
    s = fmaf(u.z, v.z, s); s = fmaf(u.w, v.w, s);
  }
  mod[t] = s + style_b[c] + 1.0f;
}

__global__ __launch_bounds__(256) void k_w2(const float* __restrict__ conv_w,
                                            float* __restrict__ W2) {
  int t = blockIdx.x * blockDim.x + threadIdx.x;
  if (t >= O_ * C_) return;
  const float* w = conv_w + (size_t)t * 9;
  float s = 0.f;
#pragma unroll
  for (int i = 0; i < 9; ++i) s = fmaf(w[i], w[i], s);
  W2[t] = s;
}

__global__ __launch_bounds__(256) void k_rnorm(const float* __restrict__ mod,
                                               const float* __restrict__ W2,
                                               float* __restrict__ rnorm) {
  int t = blockIdx.x * blockDim.x + threadIdx.x;
  if (t >= B_ * O_) return;
  int b = t >> 9, o = t & (O_ - 1);
  const float4* m4 = (const float4*)(mod + (size_t)b * C_);
  const float4* w4 = (const float4*)(W2 + (size_t)o * C_);
  float s = 0.f;
  for (int i = 0; i < C_ / 4; ++i) {
    float4 m = m4[i], w = w4[i];
    s = fmaf(m.x * m.x, w.x, s); s = fmaf(m.y * m.y, w.y, s);
    s = fmaf(m.z * m.z, w.z, s); s = fmaf(m.w * m.w, w.w, s);
  }
  rnorm[t] = 1.0f / fmaxf(sqrtf(s), 1e-12f);
}

// wq[o][j][c] = bf16(conv_w[o][c][j])   (tap-major K layout)
__global__ __launch_bounds__(256) void k_packw(const float* __restrict__ conv_w,
                                               __bf16* __restrict__ wq) {
  int t = blockIdx.x * blockDim.x + threadIdx.x;
  if (t >= O_ * KK_) return;
  int o = t / KK_;
  int r = t - o * KK_;
  int j = r >> 9;
  int c = r & (C_ - 1);
  wq[t] = (__bf16)conv_w[(size_t)(o * C_ + c) * 9 + j];
}

// xq[b][y][x][c] = bf16(imgs[b][c][y][x] * mod[b][c])   (channel-last)
__global__ __launch_bounds__(256) void k_packx(const float* __restrict__ imgs,
                                               const float* __restrict__ mod,
                                               __bf16* __restrict__ xq) {
  int t = blockIdx.x * blockDim.x + threadIdx.x;
  if (t >= B_ * HW_ * C_) return;
  int c = t & (C_ - 1);
  int pix = t >> 9;
  int b = pix >> 10;
  int yx = pix & (HW_ - 1);
  xq[t] = (__bf16)(imgs[(size_t)(b * C_ + c) * HW_ + yx] * mod[b * C_ + c]);
}

// 64-byte zero buffer for OOB async source redirection (re-zeroed every call)
__global__ void k_zero(float* __restrict__ z) {
  if (threadIdx.x < 16) z[threadIdx.x] = 0.0f;
}

// -------- main implicit-GEMM conv (bf16 WMMA + TDM + async-LDS, 2x buffered) --
// grid (N/128, M/64, B), block 128 (4 waves 2x2), wave tile 32x64

__global__ __launch_bounds__(128) void k_conv(const __bf16* __restrict__ wq,
                                              const __bf16* __restrict__ xq,
                                              const __bf16* __restrict__ zbuf,
                                              const float* __restrict__ rnorm,
                                              const float* __restrict__ conv_b,
                                              const float* __restrict__ noise,
                                              const float* __restrict__ nw,
                                              float* __restrict__ out) {
  __shared__ alignas(32) __bf16 lds_a[2][64 * LDSTRIDE];    // 64  x 32 bf16 tiles
  __shared__ alignas(32) __bf16 lds_b[2][128 * LDSTRIDE];   // 128 x 32 bf16 tiles

  const int b      = blockIdx.z;
  const int m_base = blockIdx.y * 64;
  const int p_base = blockIdx.x * 128;
  const int tid  = threadIdx.x;
  const int lane = tid & 31;
  const int wv   = tid >> 5;
  const int wm   = wv >> 1;      // 0..1 : M half
  const int wn   = wv & 1;       // 0..1 : N half

  // B staging: one thread per im2col row (pixel), 64B each
  const int py = (p_base + tid) >> 5;
  const int px = (p_base + tid) & 31;
  const uint32_t ldsAaddr[2] = { (uint32_t)(uintptr_t)&lds_a[0][0],
                                 (uint32_t)(uintptr_t)&lds_a[1][0] };
  const uint32_t ldsBaddr[2] = { (uint32_t)(uintptr_t)(&lds_b[0][0] + tid * LDSTRIDE),
                                 (uint32_t)(uintptr_t)(&lds_b[1][0] + tid * LDSTRIDE) };

  // issue the TDM weight-tile load + async im2col loads for K-chunk kc -> buffer bi
  auto issue = [&](int kc, int bi) {
    const int jj = kc >> 4;                 // conv tap 0..8 (constant per chunk)
    const int c0 = (kc & 15) << 5;          // channel base, multiple of 32
    const int dy = jj / 3 - 1;
    const int dx = jj - (jj / 3) * 3 - 1;

    // ---- A tile via Tensor Data Mover: 64 rows x 64B, row stride 9216B,
    //      LDS pad 16B per 64B row (=> 80B LDS row stride) ----
    if (wv == 0) {
      uint64_t ga = (uint64_t)(uintptr_t)(wq + (size_t)m_base * KK_ + jj * C_ + c0);
      v4u g0;
      g0[0] = 1u;                                                // count=1, user mode
      g0[1] = ldsAaddr[bi];                                      // lds_addr
      g0[2] = (uint32_t)ga;                                      // global_addr[31:0]
      g0[3] = ((uint32_t)(ga >> 32) & 0x01FFFFFFu) | (2u << 30); // addr[56:32]|type=2
      v8u g1;
      g1[0] = (2u << 16)            // data_size = 4B units
            | (1u << 20)            // pad_enable
            | (3u << 22)            // pad_interval: 16 DWORDs
            | (3u << 25);           // pad_amount : 4 DWORDs
      g1[1] = (16u << 16);          // tensor_dim0 = 16
      g1[2] = (64u << 16);          // tensor_dim1 = 64
      g1[3] = (16u << 16);          // tile_dim0 = 16
      g1[4] = 64u;                  // tile_dim1 = 64 ; tile_dim2 = 0
      g1[5] = 2304u;                // tensor_dim0_stride = 9216B / 4
      g1[6] = 0u;
      g1[7] = 0u;
      asm volatile("tensor_load_to_lds %0, %1" :: "s"(g0), "s"(g1) : "memory");
    }

    // ---- B tile via per-lane async global->LDS DMA (ASYNCcnt) ----
    {
      const int yy = py + dy;
      const int xx = px + dx;
      const bool inb = ((unsigned)yy < (unsigned)H_) && ((unsigned)xx < (unsigned)W_);
      const __bf16* src = inb
          ? (xq + ((size_t)((b * H_ + yy) * W_ + xx)) * C_ + c0)
          : zbuf;                                  // 64B of zeros for padding
      uint64_t ga = (uint64_t)(uintptr_t)src;
      const uint32_t lB = ldsBaddr[bi];
#pragma unroll
      for (int e = 0; e < 4; ++e) {
        asm volatile("global_load_async_to_lds_b128 %0, %1, off"
                     :: "v"(lB + e * 16), "v"(ga + (uint64_t)(e * 16))
                     : "memory");
      }
    }
  };

  v8f acc[2][4];
#pragma unroll
  for (int i = 0; i < 2; ++i)
#pragma unroll
    for (int jn = 0; jn < 4; ++jn)
#pragma unroll
      for (int r = 0; r < 8; ++r) acc[i][jn][r] = 0.0f;

  const int hsel = lane >> 4;    // 0: K{0-7,16-23}  1: K{8-15,24-31}
  const int lrow = lane & 15;

  // prologue: fill buffer 0
  issue(0, 0);

  for (int kc = 0; kc < NKC; ++kc) {
    const int bi = kc & 1;

    // complete the DMA that targeted buffer bi (issued last iteration / prologue)
    asm volatile("s_wait_asynccnt 0x0" ::: "memory");
    if (wv == 0) __builtin_amdgcn_s_wait_tensorcnt(0);
    __syncthreads();   // all waves: DMA visible AND done reading buffer bi^1

    // start filling the other buffer while we compute on this one
    if (kc + 1 < NKC) issue(kc + 1, bi ^ 1);

    // prefetch weights two chunks ahead (global_prefetch_b8 path)
    if (kc + 2 < NKC) {
      const int j2 = (kc + 2) >> 4;
      const int c2 = ((kc + 2) & 15) << 5;
      __builtin_prefetch(wq + (size_t)(m_base + (tid >> 1)) * KK_ + j2 * C_ + c2, 0, 1);
    }

    const __bf16* La = lds_a[bi];
    const __bf16* Lb = lds_b[bi];

    // fragment loads per ISA 16-bit A/B layouts (two ds_load_b128 each)
    Frag a[2], bb[4];
#pragma unroll
    for (int i = 0; i < 2; ++i) {
      const int r = wm * 32 + i * 16 + lrow;
      a[i].h.lo = *(const v8bf*)(La + r * LDSTRIDE + hsel * 8);
      a[i].h.hi = *(const v8bf*)(La + r * LDSTRIDE + 16 + hsel * 8);
    }
#pragma unroll
    for (int jn = 0; jn < 4; ++jn) {
      const int r = wn * 64 + jn * 16 + lrow;
      bb[jn].h.lo = *(const v8bf*)(Lb + r * LDSTRIDE + hsel * 8);
      bb[jn].h.hi = *(const v8bf*)(Lb + r * LDSTRIDE + 16 + hsel * 8);
    }

#pragma unroll
    for (int i = 0; i < 2; ++i)
#pragma unroll
      for (int jn = 0; jn < 4; ++jn)
        acc[i][jn] = __builtin_amdgcn_wmma_f32_16x16x32_bf16(
            false, a[i].v, false, bb[jn].v, (short)0, acc[i][jn], false, false);
  }

  // epilogue: demod scale + bias + noise + LeakyReLU(0.2)
  const float nwv = nw[0];
#pragma unroll
  for (int jn = 0; jn < 4; ++jn) {
    const int p = p_base + wn * 64 + jn * 16 + (lane & 15);
    const float nz = noise[b * HW_ + p] * nwv;
#pragma unroll
    for (int i = 0; i < 2; ++i) {
#pragma unroll
      for (int r = 0; r < 8; ++r) {
        const int o = m_base + wm * 32 + i * 16 + r + 8 * hsel;
        float v = acc[i][jn][r] * rnorm[b * O_ + o] + conv_b[o] + nz;
        v = (v >= 0.0f) ? v : 0.2f * v;
        out[((size_t)(b * O_ + o)) * HW_ + p] = v;
      }
    }
  }
}

// ---------------- launcher ----------------

extern "C" void kernel_launch(void* const* d_in, const int* in_sizes, int n_in,
                              void* d_out, int out_size, void* d_ws, size_t ws_size,
                              hipStream_t stream) {
  const float* imgs    = (const float*)d_in[0];
  const float* w_embs  = (const float*)d_in[1];
  const float* noise   = (const float*)d_in[2];
  const float* conv_w  = (const float*)d_in[3];
  const float* conv_b  = (const float*)d_in[4];
  const float* style_w = (const float*)d_in[5];
  const float* style_b = (const float*)d_in[6];
  const float* nw      = (const float*)d_in[7];
  float* out = (float*)d_out;

  char* ws = (char*)d_ws;
  float*  mod   = (float*)(ws + 0);                      //  32 KB
  float*  W2    = (float*)(ws + 32768);                  //   1 MB
  float*  rnorm = (float*)(ws + 32768 + 1048576);        //  32 KB
  __bf16* wq    = (__bf16*)(ws + 1114112);               // 4.72 MB
  __bf16* xq    = (__bf16*)(ws + 5832704);               // 16.8 MB
  float*  zbuf  = (float*)(ws + 22609920);               //   64 B of zeros

  k_style<<<(B_ * C_) / 256, 256, 0, stream>>>(w_embs, style_w, style_b, mod);
  k_w2<<<(O_ * C_) / 256, 256, 0, stream>>>(conv_w, W2);
  k_rnorm<<<(B_ * O_) / 256, 256, 0, stream>>>(mod, W2, rnorm);
  k_packw<<<(O_ * KK_) / 256, 256, 0, stream>>>(conv_w, wq);
  k_packx<<<(B_ * HW_ * C_) / 256, 256, 0, stream>>>(imgs, mod, xq);
  k_zero<<<1, 32, 0, stream>>>(zbuf);

  dim3 grid(HW_ / 128, O_ / 64, B_);
  k_conv<<<grid, 128, 0, stream>>>(wq, xq, (const __bf16*)zbuf, rnorm, conv_b,
                                   noise, nw, out);
}
